// WindowAttention_51659866636468
// MI455X (gfx1250) — compile-verified, matrix operand order checked by reading, not run on hardware
//
#include <hip/hip_runtime.h>

// ---------------------------------------------------------------------------
// Window attention (Swin-style), fused per-window, bf16 WMMA on gfx1250.
// N=64 tokens, C=128, H=4 heads, D=32. One workgroup (8 wave32) per window.
// ---------------------------------------------------------------------------

typedef __attribute__((ext_vector_type(16))) __bf16 v16bf;
typedef __attribute__((ext_vector_type(8)))  float  v8f;

#define DEVI __device__ __forceinline__

constexpr int NTOK = 64;    // tokens per window
constexpr int CH   = 128;   // channels
constexpr int SX   = 136;   // LDS row stride (bf16) for 128-wide tiles (pad -> conflict-free b128 frag loads)
constexpr int SVT  = 72;    // LDS row stride for V^T (64 tokens wide)
constexpr int SP   = 72;    // LDS row stride for probabilities (64 wide)

// dynamic LDS layout (bytes, all 16B aligned)
constexpr int OFF_X  = 0;                       // 64  x SX bf16 = 17408   (also reused as sO)
constexpr int OFF_Q  = 17408;                   // 64  x SX bf16 = 17408
constexpr int OFF_K  = 34816;                   // 64  x SX bf16 = 17408
constexpr int OFF_VT = 52224;                   // 128 x SVT bf16 = 18432  (V transposed: [feat][token])
constexpr int OFF_P  = 70656;                   // 256 x SP bf16 = 36864   ([h*64+m][token])
constexpr int SMEM_BYTES = 107520;

struct F2 { uint4 lo, hi; };

DEVI unsigned short f2bf(float f) {             // f32 -> bf16, round-to-nearest-even (scalar fallback)
  unsigned int u = __builtin_bit_cast(unsigned int, f);
  u += 0x7FFFu + ((u >> 16) & 1u);
  return (unsigned short)(u >> 16);
}

// Packed f32x2 -> bf16x2 in one VALU op. The builtin is gated off for gfx1250 in
// clang, but the instruction is CDNA-lineage; probe it through the assembler.
DEVI unsigned int f2bf_pk(float a, float b) {
  unsigned int r;
  asm("v_cvt_pk_bf16_f32 %0, %1, %2" : "=v"(r) : "v"(a), "v"(b));
  return r;
}

// Load one 16x32 bf16 WMMA operand fragment (A-layout; pass B^T for the B operand)
// from a row-major [row][k] array. Per ISA: lanes 0-15 hold rows, k {0..7,16..23};
// lanes 16-31 hold rows, k {8..15,24..31}.  Two 16B loads per lane.
DEVI v16bf load_frag(const unsigned short* base, int stride, int row0, int k0) {
  const int lane = threadIdx.x & 31;
  const unsigned short* p = base + (row0 + (lane & 15)) * stride + k0 + ((lane >> 4) << 3);
  F2 f;
  f.lo = *reinterpret_cast<const uint4*>(p);        // k .. k+7
  f.hi = *reinterpret_cast<const uint4*>(p + 16);   // k+16 .. k+23
  return __builtin_bit_cast(v16bf, f);
}

DEVI v8f wmma_bf16(v16bf a, v16bf b, v8f c) {
  return __builtin_amdgcn_wmma_f32_16x16x32_bf16(false, a, false, b, (short)0, c, false, false);
}

// ---------------------------------------------------------------------------
// Prep: convert weights to bf16 in workspace; gather relative-position bias.
// ---------------------------------------------------------------------------
__global__ void prep_kernel(const float* __restrict__ qkv_w, const float* __restrict__ proj_w,
                            const float* __restrict__ bias_table, const int* __restrict__ rel_idx,
                            unsigned short* __restrict__ wqkv, unsigned short* __restrict__ wproj,
                            float* __restrict__ rpb) {
  int i = blockIdx.x * blockDim.x + threadIdx.x;
  if (i < 49152) {                              // 384x128 qkv weights
    wqkv[i] = f2bf(qkv_w[i]);
  } else if (i < 65536) {                       // 128x128 proj weights
    int j = i - 49152;
    wproj[j] = f2bf(proj_w[j]);
  } else if (i < 81920) {                       // rpb[h][m][n] = bias_table[rel_idx[m,n], h]
    int j = i - 65536;                          // h*4096 + m*64 + n
    int h = j >> 12, mn = j & 4095;
    rpb[j] = bias_table[rel_idx[mn] * 4 + h];
  }
}

// ---------------------------------------------------------------------------
// Fused window attention. gridDim.x = #windows. blockDim = 256 (8 wave32).
// ---------------------------------------------------------------------------
__global__ __launch_bounds__(256) void win_attn_kernel(
    const float* __restrict__ x, const float* __restrict__ maskp,
    const float* __restrict__ qkvb, const float* __restrict__ projb,
    const unsigned short* __restrict__ wqkv, const unsigned short* __restrict__ wproj,
    const float* __restrict__ rpb, float* __restrict__ out, int nW) {
  extern __shared__ __align__(16) unsigned char smem[];
  unsigned short* sX  = (unsigned short*)(smem + OFF_X);
  unsigned short* sQ  = (unsigned short*)(smem + OFF_Q);
  unsigned short* sK  = (unsigned short*)(smem + OFF_K);
  unsigned short* sVt = (unsigned short*)(smem + OFF_VT);
  unsigned short* sP  = (unsigned short*)(smem + OFF_P);

  const int b    = blockIdx.x;
  const int tid  = threadIdx.x;
  // wave id as an SGPR: scalar tile indices -> s_cbranch, SALU addressing,
  // EXEC provably all-ones around WMMA (ISA requirement).
  const int w    = __builtin_amdgcn_readfirstlane(tid >> 5);   // 0..7
  const int lane = tid & 31;
  const int hi   = lane >> 4;       // C-tile row half (rows r / r+8)
  const int ln   = lane & 15;       // C-tile column within tile

  // warm L2 for this block's slice of the shared weights
  __builtin_prefetch(wqkv + w * 6144, 0, 1);

  // ---- Phase A: load x tile (64x128 f32), convert to bf16 into sX ----------
  {
    const float4* x4 = (const float4*)(x + (size_t)b * (NTOK * CH));
    #pragma unroll
    for (int j = 0; j < 8; ++j) {
      int idx4 = tid + j * 256;                 // 0..2047 float4s
      float4 v = x4[idx4];
      int row = idx4 >> 5;
      int col = (idx4 & 31) << 2;
      uint2 pk;
      pk.x = f2bf_pk(v.x, v.y);
      pk.y = f2bf_pk(v.z, v.w);
      *reinterpret_cast<uint2*>(sX + row * SX + col) = pk;
    }
  }
  __syncthreads();

  // ---- Phase B: QKV = X(64x128) @ Wqkv^T(128x384) + b -----------------------
  // Q scaled & stored row-major; K row-major; V stored TRANSPOSED ([feat][tok]).
  {
    const int mt = w & 3;                       // fixed M-tile per wave (scalar)
    const int ntBase = (w >> 2) * 12;           // 12 N-tiles per wave (scalar)
    v16bf aF[4];
    #pragma unroll
    for (int kk = 0; kk < 4; ++kk) aF[kk] = load_frag(sX, SX, mt * 16, kk * 32);

    const int tokBase = mt * 16 + hi * 8;
    for (int t = 0; t < 12; ++t) {
      const int nt = ntBase + t;                // scalar
      v8f acc = {};
      #pragma unroll
      for (int kk = 0; kk < 4; ++kk) {
        v16bf bF = load_frag(wqkv, CH, nt * 16, kk * 32);   // W rows == B^T rows
        acc = wmma_bf16(aF[kk], bF, acc);
      }
      const int ng = nt * 16 + ln;              // global output column 0..383
      const float bias = qkvb[ng];
      #pragma unroll
      for (int r = 0; r < 8; ++r) acc[r] += bias;
      if (nt < 8) {                             // Q (cols 0..127), pre-scaled by D^-0.5
        #pragma unroll
        for (int r = 0; r < 8; r += 2) {
          unsigned int u = f2bf_pk(acc[r]     * 0.17677669529663687f,
                                   acc[r + 1] * 0.17677669529663687f);
          sQ[(tokBase + r)     * SX + ng] = (unsigned short)u;
          sQ[(tokBase + r + 1) * SX + ng] = (unsigned short)(u >> 16);
        }
      } else if (nt < 16) {                     // K (cols 128..255)
        #pragma unroll
        for (int r = 0; r < 8; r += 2) {
          unsigned int u = f2bf_pk(acc[r], acc[r + 1]);
          sK[(tokBase + r)     * SX + (ng - 128)] = (unsigned short)u;
          sK[(tokBase + r + 1) * SX + (ng - 128)] = (unsigned short)(u >> 16);
        }
      } else {                                  // V (cols 256..383): transposed, contiguous in r
        uint4 pk;
        pk.x = f2bf_pk(acc[0], acc[1]);
        pk.y = f2bf_pk(acc[2], acc[3]);
        pk.z = f2bf_pk(acc[4], acc[5]);
        pk.w = f2bf_pk(acc[6], acc[7]);
        *reinterpret_cast<uint4*>(sVt + (ng - 256) * SVT + tokBase) = pk;
      }
    }
  }
  __syncthreads();

  // ---- Phase C: S = Q·K^T + rpb + mask ; row softmax in registers -> sP -----
  {
    const int wIdx = ((nW & (nW - 1)) == 0) ? (b & (nW - 1)) : (b % nW);
    const float* maskw = maskp + (size_t)wIdx * (NTOK * NTOK);
    for (int rb = w; rb < 16; rb += 8) {        // row-block = (head, M-tile), scalar
      const int h = rb >> 2, mt = rb & 3;
      v16bf aF = load_frag(sQ, SX, mt * 16, h * 32);        // K-dim = D = 32: one step
      v8f acc[4];
      // initialize accumulators with rpb + mask (folds the bias add into WMMA's C input)
      #pragma unroll
      for (int nt = 0; nt < 4; ++nt) {
        const int n = nt * 16 + ln;
        #pragma unroll
        for (int r = 0; r < 8; ++r) {
          const int m = mt * 16 + hi * 8 + r;
          acc[nt][r] = rpb[(h * 64 + m) * 64 + n] + maskw[m * 64 + n];
        }
      }
      #pragma unroll
      for (int nt = 0; nt < 4; ++nt) {
        v16bf bF = load_frag(sK, SX, nt * 16, h * 32);
        acc[nt] = wmma_bf16(aF, bF, acc[nt]);
      }
      // softmax over the 64-wide row: reduce across 4 tiles x 16 lanes (half-wave)
      float inv[8];
      #pragma unroll
      for (int r = 0; r < 8; ++r) {
        float mx = fmaxf(fmaxf(acc[0][r], acc[1][r]), fmaxf(acc[2][r], acc[3][r]));
        mx = fmaxf(mx, __shfl_xor(mx, 1, 32));
        mx = fmaxf(mx, __shfl_xor(mx, 2, 32));
        mx = fmaxf(mx, __shfl_xor(mx, 4, 32));
        mx = fmaxf(mx, __shfl_xor(mx, 8, 32));
        float s = 0.0f;
        #pragma unroll
        for (int nt = 0; nt < 4; ++nt) { acc[nt][r] = __expf(acc[nt][r] - mx); s += acc[nt][r]; }
        s += __shfl_xor(s, 1, 32);
        s += __shfl_xor(s, 2, 32);
        s += __shfl_xor(s, 4, 32);
        s += __shfl_xor(s, 8, 32);
        inv[r] = 1.0f / s;
      }
      #pragma unroll
      for (int nt = 0; nt < 4; ++nt) {
        const int n = nt * 16 + ln;
        #pragma unroll
        for (int r = 0; r < 8; r += 2) {
          const int m = mt * 16 + hi * 8 + r;
          unsigned int u = f2bf_pk(acc[nt][r] * inv[r], acc[nt][r + 1] * inv[r + 1]);
          sP[(h * 64 + m)     * SP + n] = (unsigned short)u;
          sP[(h * 64 + m + 1) * SP + n] = (unsigned short)(u >> 16);
        }
      }
    }
  }
  __syncthreads();

  // ---- Phase D: O = P(64x64) @ V(64x32) per head -> sO (reuses sX) ----------
  {
    unsigned short* sO = sX;
    const int mt = w >> 1, dt = w & 1;          // fixed (M-tile, D-tile) per wave (scalar)
    const int tokBase = mt * 16 + hi * 8;
    #pragma unroll
    for (int h = 0; h < 4; ++h) {
      v8f acc = {};
      #pragma unroll
      for (int kk = 0; kk < 2; ++kk) {          // K = 64 tokens = 2 steps
        v16bf aF = load_frag(sP,  SP,  h * 64 + mt * 16, kk * 32);
        v16bf bF = load_frag(sVt, SVT, h * 32 + dt * 16, kk * 32);  // rows = feats (V^T)
        acc = wmma_bf16(aF, bF, acc);
      }
      const int feat = h * 32 + dt * 16 + ln;
      #pragma unroll
      for (int r = 0; r < 8; r += 2) {
        unsigned int u = f2bf_pk(acc[r], acc[r + 1]);
        sO[(tokBase + r)     * SX + feat] = (unsigned short)u;
        sO[(tokBase + r + 1) * SX + feat] = (unsigned short)(u >> 16);
      }
    }
  }
  __syncthreads();

  // ---- Phase E: out = O(64x128) @ proj_w^T(128x128) + proj_b ----------------
  {
    const unsigned short* sO = sX;
    float* outB = out + (size_t)b * (NTOK * CH);
    const int mt = w & 3;                       // fixed M-tile per wave (scalar)
    v16bf aF[4];
    #pragma unroll
    for (int kk = 0; kk < 4; ++kk) aF[kk] = load_frag(sO, SX, mt * 16, kk * 32);
    const int tokBase = mt * 16 + hi * 8;
    #pragma unroll
    for (int t = 0; t < 4; ++t) {
      const int nt2 = (w >> 2) + 2 * t;         // 8 N-tiles across waves (scalar)
      v8f acc = {};
      #pragma unroll
      for (int kk = 0; kk < 4; ++kk) {
        v16bf bF = load_frag(wproj, CH, nt2 * 16, kk * 32);
        acc = wmma_bf16(aF[kk], bF, acc);
      }
      const int ng = nt2 * 16 + ln;
      const float pb = projb[ng];
      #pragma unroll
      for (int r = 0; r < 8; ++r)
        outB[(tokBase + r) * CH + ng] = acc[r] + pb;
    }
  }
}

// ---------------------------------------------------------------------------
extern "C" void kernel_launch(void* const* d_in, const int* in_sizes, int n_in,
                              void* d_out, int out_size, void* d_ws, size_t ws_size,
                              hipStream_t stream) {
  const float* x          = (const float*)d_in[0];
  const float* mask       = (const float*)d_in[1];
  const float* bias_table = (const float*)d_in[2];
  const float* qkv_w      = (const float*)d_in[3];
  const float* qkv_b      = (const float*)d_in[4];
  const float* proj_w     = (const float*)d_in[5];
  const float* proj_b     = (const float*)d_in[6];
  const int*   rel_idx    = (const int*)d_in[7];

  const int B  = in_sizes[0] / (NTOK * CH);     // 8192 windows
  const int nW = in_sizes[1] / (NTOK * NTOK);   // 4096 masks

  // workspace: [0, 98304) qkv_w bf16 | [98304, 131072) proj_w bf16 | [131072, 196608) rpb f32
  unsigned short* wqkv  = (unsigned short*)d_ws;
  unsigned short* wproj = wqkv + 49152;
  float*          rpb   = (float*)((char*)d_ws + 131072);

  prep_kernel<<<320, 256, 0, stream>>>(qkv_w, proj_w, bias_table, rel_idx, wqkv, wproj, rpb);
  win_attn_kernel<<<B, 256, SMEM_BYTES, stream>>>(x, mask, qkv_b, proj_b, wqkv, wproj,
                                                  rpb, (float*)d_out, nW);
}
